// Critic_20418274525541
// MI455X (gfx1250) — matrix operand order repaired and checked by reading, not compiled, not run on hardware
//
#include <hip/hip_runtime.h>
#include <math.h>

// Problem constants (match reference)
#define NNODES 50000
#define NEDGES 600000
#define CIN    32
#define HID    128
#define EDIM   16
#define EPAD   32                  // ea stored padded to 32 cols (16..31 zero)
#define NG     64
#define EN     (NEDGES + NNODES)   // 650000, divisible by 16
#define NEGS   0.2f

typedef __attribute__((ext_vector_type(16))) _Float16 v16h;
typedef __attribute__((ext_vector_type(8)))  float    v8f;
typedef __attribute__((ext_vector_type(4)))  float    v4f;

// ---------------------------------------------------------------------------
// Weight pack: B f32 [K,128] row-major -> Bt f16 [128, Kpad] (transposed,
// zero padded). Bt[c*Kpad + k] = k < K ? B[k*128 + c] : 0
// ---------------------------------------------------------------------------
__global__ void k_pack_w(const float* __restrict__ B, _Float16* __restrict__ Bt,
                         int K, int Kpad) {
  int i = blockIdx.x * blockDim.x + threadIdx.x;
  if (i >= HID * Kpad) return;
  int c = i / Kpad, k = i - c * Kpad;
  Bt[i] = (k < K) ? (_Float16)B[(size_t)k * HID + c] : (_Float16)0.f;
}

// ---------------------------------------------------------------------------
// WMMA GEMM: C[M,128] = act( A[M,K] @ W + bias ), W pre-packed f16 [128,K]^T.
// K multiple of 32. A row-major f32 [M, lda], lda multiple of 4.
// Block = 256 threads = 8 waves; wave w owns cols [16w,16w+16), block owns
// rows [16*blockIdx.x, +16). Unguarded 128-bit loads; A/B k order
// contiguous, k = 16*(lane/16) + s.
// ---------------------------------------------------------------------------
__global__ __launch_bounds__(256) void k_gemm_wmma(
    const float* __restrict__ A, int lda,
    const _Float16* __restrict__ Bt,   // [128, K] f16 (transposed weights)
    const float* __restrict__ bias,    // may be nullptr
    float* __restrict__ C,
    int K, int act)                    // act: 0 = none, 1 = tanh
{
  const int lane  = threadIdx.x & 31;
  const int wave  = threadIdx.x >> 5;
  const int mtile = blockIdx.x * 16;
  const int r16   = lane & 15;
  const bool hi   = lane >= 16;
  const int col   = wave * 16 + r16;

  const float*    Ar = A  + (size_t)(mtile + r16) * lda + (hi ? 16 : 0);
  const _Float16* Br = Bt + (size_t)col * K + (hi ? 16 : 0);

  v8f acc = {};
  for (int k0 = 0; k0 < K; k0 += 32) {
    // ---- A fragment: one contiguous 16-float run (K = kb..kb+15)
    v4f f0 = *(const v4f*)(Ar + k0);
    v4f f1 = *(const v4f*)(Ar + k0 + 4);
    v4f f2 = *(const v4f*)(Ar + k0 + 8);
    v4f f3 = *(const v4f*)(Ar + k0 + 12);
    v16h a;
#pragma unroll
    for (int j = 0; j < 4; ++j) {
      a[j]      = (_Float16)f0[j];
      a[4 + j]  = (_Float16)f1[j];
      a[8 + j]  = (_Float16)f2[j];
      a[12 + j] = (_Float16)f3[j];
    }
    // ---- B fragment: 16 contiguous halves of this lane's weight column
    v16h b = *(const v16h*)(Br + k0);
    acc = __builtin_amdgcn_wmma_f32_16x16x32_f16(false, a, false, b,
                                                 (short)0, acc, false, false);
  }
  // ---- store: VGPR i -> row mtile+i (+8 for hi lanes), this lane's col
  const float bv = bias ? bias[col] : 0.f;
#pragma unroll
  for (int i = 0; i < 8; ++i) {
    int row = mtile + i + (hi ? 8 : 0);
    float v = acc[i] + bv;
    if (act) v = tanhf(v);
    C[(size_t)row * HID + col] = v;
  }
}

// ---------------------------------------------------------------------------
// Scalar helpers
// ---------------------------------------------------------------------------
__device__ __forceinline__ void atomicMaxFloat(float* addr, float v) {
  if (v >= 0.f) atomicMax((int*)addr, __float_as_int(v));
  else          atomicMin((unsigned int*)addr, (unsigned int)__float_as_int(v));
}

__global__ void k_fill(float* p, float v, int n) {
  int i = blockIdx.x * blockDim.x + threadIdx.x;
  if (i < n) p[i] = v;
}

__global__ void k_init_bias(float* out, const float* __restrict__ bias, int rows) {
  int i = blockIdx.x * blockDim.x + threadIdx.x;
  if (i < rows * HID) out[i] = bias[i & (HID - 1)];
}

// deg[d] += 1 ; ea[E+d][c] += edge_attr[e][c] ; ea[e][c] = edge_attr[e][c]
__global__ void k_deg_loop(const int* __restrict__ ei, const float* __restrict__ eattr,
                           float* __restrict__ ea, float* __restrict__ deg) {
  int e = blockIdx.x * blockDim.x + threadIdx.x;
  if (e >= NEDGES) return;
  int d = ei[NEDGES + e];
  atomicAdd(&deg[d], 1.0f);
#pragma unroll
  for (int c = 0; c < EDIM; ++c) {
    float v = eattr[(size_t)e * EDIM + c];
    ea[(size_t)e * EPAD + c] = v;
    atomicAdd(&ea[(size_t)(NEDGES + d) * EPAD + c], v);
  }
}

__global__ void k_loop_div(float* ea, const float* __restrict__ deg) {
  int i = blockIdx.x * blockDim.x + threadIdx.x;
  if (i >= NNODES * EDIM) return;
  int n = i / EDIM, c = i - n * EDIM;
  ea[(size_t)(NEDGES + n) * EPAD + c] /= fmaxf(deg[n], 1.0f);
}

// wave per edge: z = leaky(xl[s]+xr[d]+e); logit = z . att ; segment max
__global__ void k_edge_logit(const int* __restrict__ ei,
                             const float* __restrict__ xl, const float* __restrict__ xr,
                             const float* __restrict__ e,  const float* __restrict__ att,
                             float* __restrict__ logit, float* __restrict__ mmax) {
  int t = blockIdx.x * blockDim.x + threadIdx.x;
  int idx = t >> 5, lane = t & 31;
  if (idx >= EN) return;
  int s = (idx < NEDGES) ? ei[idx]          : idx - NEDGES;
  int d = (idx < NEDGES) ? ei[NEDGES + idx] : idx - NEDGES;
  float p = 0.f;
#pragma unroll
  for (int j = 0; j < 4; ++j) {
    int c = lane + 32 * j;
    float z = xl[(size_t)s * HID + c] + xr[(size_t)d * HID + c] + e[(size_t)idx * HID + c];
    z = (z > 0.f) ? z : NEGS * z;
    p += z * att[c];
  }
  for (int off = 16; off; off >>= 1) p += __shfl_xor(p, off);
  if (lane == 0) { logit[idx] = p; atomicMaxFloat(&mmax[d], p); }
}

__global__ void k_edge_w(const int* __restrict__ ei, const float* __restrict__ logit,
                         const float* __restrict__ m, float* __restrict__ w,
                         float* __restrict__ denom) {
  int idx = blockIdx.x * blockDim.x + threadIdx.x;
  if (idx >= EN) return;
  int d = (idx < NEDGES) ? ei[NEDGES + idx] : idx - NEDGES;
  float v = __expf(logit[idx] - m[d]);
  w[idx] = v;
  atomicAdd(&denom[d], v);
}

// wave per edge: hout[d] += alpha * xl[s]
__global__ void k_edge_agg(const int* __restrict__ ei, const float* __restrict__ w,
                           const float* __restrict__ denom, const float* __restrict__ xl,
                           float* __restrict__ hout) {
  int t = blockIdx.x * blockDim.x + threadIdx.x;
  int idx = t >> 5, lane = t & 31;
  if (idx >= EN) return;
  int s = (idx < NEDGES) ? ei[idx]          : idx - NEDGES;
  int d = (idx < NEDGES) ? ei[NEDGES + idx] : idx - NEDGES;
  float alpha = w[idx] / denom[d];
#pragma unroll
  for (int j = 0; j < 4; ++j) {
    int c = lane + 32 * j;
    atomicAdd(&hout[(size_t)d * HID + c], alpha * xl[(size_t)s * HID + c]);
  }
}

__global__ void k_tanh(const float* __restrict__ in, float* __restrict__ out, int n) {
  int i = blockIdx.x * blockDim.x + threadIdx.x;
  if (i < n) out[i] = tanhf(in[i]);
}

// wave per row: out[r] = act( A[r,:] . w + b )
__global__ void k_rowdot(const float* __restrict__ A, const float* __restrict__ w,
                         const float* __restrict__ b, float* __restrict__ out,
                         int rows, int act) {
  int t = blockIdx.x * blockDim.x + threadIdx.x;
  int r = t >> 5, lane = t & 31;
  if (r >= rows) return;
  float p = 0.f;
#pragma unroll
  for (int j = 0; j < 4; ++j) { int c = lane + 32 * j; p += A[(size_t)r * HID + c] * w[c]; }
  for (int off = 16; off; off >>= 1) p += __shfl_xor(p, off);
  if (lane == 0) { float v = p + b[0]; if (act) v = tanhf(v); out[r] = v; }
}

__global__ void k_pool_max(const int* __restrict__ bid, const float* __restrict__ g,
                           float* __restrict__ gm) {
  int n = blockIdx.x * blockDim.x + threadIdx.x;
  if (n < NNODES) atomicMaxFloat(&gm[bid[n]], g[n]);
}
__global__ void k_pool_w(const int* __restrict__ bid, const float* __restrict__ g,
                         const float* __restrict__ gm, float* __restrict__ gw,
                         float* __restrict__ gden) {
  int n = blockIdx.x * blockDim.x + threadIdx.x;
  if (n >= NNODES) return;
  float v = __expf(g[n] - gm[bid[n]]);
  gw[n] = v;
  atomicAdd(&gden[bid[n]], v);
}
__global__ void k_pool_agg(const int* __restrict__ bid, const float* __restrict__ gw,
                           const float* __restrict__ gden, const float* __restrict__ hn,
                           float* __restrict__ pooled) {
  int t = blockIdx.x * blockDim.x + threadIdx.x;
  int n = t >> 5, lane = t & 31;
  if (n >= NNODES) return;
  int b = bid[n];
  float coef = gw[n] / gden[b];
#pragma unroll
  for (int j = 0; j < 4; ++j) {
    int c = lane + 32 * j;
    atomicAdd(&pooled[(size_t)b * HID + c], coef * hn[(size_t)n * HID + c]);
  }
}

// ---------------------------------------------------------------------------
// Host orchestration
// ---------------------------------------------------------------------------
extern "C" void kernel_launch(void* const* d_in, const int* in_sizes, int n_in,
                              void* d_out, int out_size, void* d_ws, size_t ws_size,
                              hipStream_t stream) {
  (void)n_in; (void)out_size; (void)ws_size;
  const float* x     = (const float*)d_in[0];
  const int*   ei    = (const int*)d_in[1];
  const float* eattr = (const float*)d_in[2];
  const int*   bid   = (const int*)d_in[3];

  // ---- unpack params; handle both pytree flatten orders ----
  const float *gWe[5], *gWl[5], *gWr[5], *gatt[5], *gbias[5], *gbl[5], *gbr[5];
  const float *gateW[3], *gateB[3], *featW[2], *featB[2], *fcW[3], *fcB[3];
  int p = 4;
  auto nxt = [&]() { return (const float*)d_in[p++]; };
  if (in_sizes[4] == HID * HID) {
    // sorted-key pytree order: fc, feat, gate, gats{We,Wl,Wr,att,bias,bl,br}
    for (int i = 0; i < 3; ++i) { fcW[i] = nxt(); fcB[i] = nxt(); }
    for (int i = 0; i < 2; ++i) { featW[i] = nxt(); featB[i] = nxt(); }
    for (int i = 0; i < 3; ++i) { gateW[i] = nxt(); gateB[i] = nxt(); }
    for (int l = 0; l < 5; ++l) {
      gWe[l] = nxt(); gWl[l] = nxt(); gWr[l] = nxt();
      gatt[l] = nxt(); gbias[l] = nxt(); gbl[l] = nxt(); gbr[l] = nxt();
    }
  } else {
    // insertion order: gats{Wl,bl,Wr,br,We,att,bias}, gate, feat, fc
    for (int l = 0; l < 5; ++l) {
      gWl[l] = nxt(); gbl[l] = nxt(); gWr[l] = nxt(); gbr[l] = nxt();
      gWe[l] = nxt(); gatt[l] = nxt(); gbias[l] = nxt();
    }
    for (int i = 0; i < 3; ++i) { gateW[i] = nxt(); gateB[i] = nxt(); }
    for (int i = 0; i < 2; ++i) { featW[i] = nxt(); featB[i] = nxt(); }
    for (int i = 0; i < 3; ++i) { fcW[i] = nxt(); fcB[i] = nxt(); }
  }

  // ---- carve workspace ----
  char* wsb = (char*)d_ws;
  size_t off = 0;
  auto alloc = [&](size_t nfloats) {
    float* r = (float*)(wsb + off);
    off += ((nfloats * sizeof(float)) + 255) & ~(size_t)255;
    return r;
  };
  float* deg    = alloc(NNODES);
  float* ea     = alloc((size_t)EN * EPAD);       // padded to 32 cols
  float* h      = alloc((size_t)NNODES * HID);
  float* xl     = alloc((size_t)NNODES * HID);    // reused as MLP temp 1
  float* xr     = alloc((size_t)NNODES * HID);    // reused as MLP temp 2
  float* ebuf   = alloc((size_t)EN * HID);
  float* logit  = alloc(EN);                      // reused as g[N]
  float* wv     = alloc(EN);                      // reused as gw[N]
  float* mmax   = alloc(NNODES);
  float* denom  = alloc(NNODES);
  float* hout   = alloc((size_t)NNODES * HID);    // reused as hn
  float* gm     = alloc(NG);
  float* gden   = alloc(NG);
  float* pooled = alloc((size_t)NG * HID);
  float* f1     = alloc((size_t)NG * HID);
  float* f2     = alloc((size_t)NG * HID);
  _Float16* wt1 = (_Float16*)alloc(HID * HID / 2);  // packed weight scratch
  _Float16* wt2 = (_Float16*)alloc(HID * HID / 2);
  _Float16* wt3 = (_Float16*)alloc(HID * HID / 2);

  const int BT = 256;
  auto cdiv = [](long long a, long long b) { return (unsigned)((a + b - 1) / b); };
  auto pack = [&](const float* W, _Float16* Bt, int K, int Kpad) {
    k_pack_w<<<cdiv((size_t)HID * Kpad, BT), BT, 0, stream>>>(W, Bt, K, Kpad);
  };

  // ---- self-loop attrs: ea = concat(edge_attr, segsum(edge_attr,dst)/deg) ----
  k_fill<<<cdiv(NNODES, BT), BT, 0, stream>>>(deg, 0.f, NNODES);
  k_fill<<<cdiv((size_t)EN * EPAD, BT), BT, 0, stream>>>(ea, 0.f, EN * EPAD);
  k_deg_loop<<<cdiv(NEDGES, BT), BT, 0, stream>>>(ei, eattr, ea, deg);
  k_loop_div<<<cdiv((size_t)NNODES * EDIM, BT), BT, 0, stream>>>(ea, deg);

  // ---- 5 GATv2 layers ----
  for (int l = 0; l < 5; ++l) {
    const int din = (l == 0) ? CIN : HID;
    const float* hin = (l == 0) ? x : h;
    pack(gWl[l], wt1, din, din);
    pack(gWr[l], wt2, din, din);
    pack(gWe[l], wt3, EDIM, EPAD);
    k_gemm_wmma<<<dim3(NNODES / 16), BT, 0, stream>>>(hin, din, wt1, gbl[l], xl, din, 0);
    k_gemm_wmma<<<dim3(NNODES / 16), BT, 0, stream>>>(hin, din, wt2, gbr[l], xr, din, 0);
    k_gemm_wmma<<<dim3(EN / 16),     BT, 0, stream>>>(ea, EPAD, wt3, nullptr, ebuf, EPAD, 0);

    k_fill<<<cdiv(NNODES, BT), BT, 0, stream>>>(mmax, -INFINITY, NNODES);
    k_fill<<<cdiv(NNODES, BT), BT, 0, stream>>>(denom, 0.f, NNODES);
    k_init_bias<<<cdiv((size_t)NNODES * HID, BT), BT, 0, stream>>>(hout, gbias[l], NNODES);

    k_edge_logit<<<cdiv((size_t)EN * 32, BT), BT, 0, stream>>>(ei, xl, xr, ebuf, gatt[l], logit, mmax);
    k_edge_w<<<cdiv(EN, BT), BT, 0, stream>>>(ei, logit, mmax, wv, denom);
    k_edge_agg<<<cdiv((size_t)EN * 32, BT), BT, 0, stream>>>(ei, wv, denom, xl, hout);
    k_tanh<<<cdiv((size_t)NNODES * HID, BT), BT, 0, stream>>>(hout, h, NNODES * HID);
  }

  // ---- gate_nn: g = tanh(tanh(h W1+b1) W2+b2) W3 + b3 ----
  pack(gateW[0], wt1, HID, HID);
  pack(gateW[1], wt2, HID, HID);
  k_gemm_wmma<<<dim3(NNODES / 16), BT, 0, stream>>>(h,  HID, wt1, gateB[0], xl, HID, 1);
  k_gemm_wmma<<<dim3(NNODES / 16), BT, 0, stream>>>(xl, HID, wt2, gateB[1], xr, HID, 1);
  k_rowdot<<<cdiv((size_t)NNODES * 32, BT), BT, 0, stream>>>(xr, gateW[2], gateB[2], logit, NNODES, 0);

  // ---- feat nn: hn = tanh(tanh(h Wn1+bn1) Wn2+bn2) ----
  pack(featW[0], wt1, HID, HID);
  pack(featW[1], wt2, HID, HID);
  k_gemm_wmma<<<dim3(NNODES / 16), BT, 0, stream>>>(h,  HID, wt1, featB[0], xl, HID, 1);
  k_gemm_wmma<<<dim3(NNODES / 16), BT, 0, stream>>>(xl, HID, wt2, featB[1], hout, HID, 1);

  // ---- segment softmax over batch + weighted pool ----
  k_fill<<<1, NG, 0, stream>>>(gm, -INFINITY, NG);
  k_fill<<<1, NG, 0, stream>>>(gden, 0.f, NG);
  k_fill<<<cdiv((size_t)NG * HID, BT), BT, 0, stream>>>(pooled, 0.f, NG * HID);
  k_pool_max<<<cdiv(NNODES, BT), BT, 0, stream>>>(bid, logit, gm);
  k_pool_w<<<cdiv(NNODES, BT), BT, 0, stream>>>(bid, logit, gm, wv, gden);
  k_pool_agg<<<cdiv((size_t)NNODES * 32, BT), BT, 0, stream>>>(bid, wv, gden, hout, pooled);

  // ---- fc head ----
  pack(fcW[0], wt1, HID, HID);
  pack(fcW[1], wt2, HID, HID);
  k_gemm_wmma<<<dim3(NG / 16), BT, 0, stream>>>(pooled, HID, wt1, fcB[0], f1, HID, 1);
  k_gemm_wmma<<<dim3(NG / 16), BT, 0, stream>>>(f1,     HID, wt2, fcB[1], f2, HID, 1);
  k_rowdot<<<cdiv((size_t)NG * 32, BT), BT, 0, stream>>>(f2, fcW[2], fcB[2], (float*)d_out, NG, 1);
}
